// FlatSparseAttention_19713899889031
// MI455X (gfx1250) — matrix operand order repaired and checked by reading, not compiled, not run on hardware
//
#include <hip/hip_runtime.h>
#include <hip/hip_bf16.h>

// ---------------------------------------------------------------------------
// FlatSparseAttention for MI455X (gfx1250, wave32, WMMA)
//   scores = (x @ W^T + b) / max(T, 0.5)        [65536 x 576]
//   attn   = sparsemax(scores)  (exact sort-based threshold, per reference)
//   out    = (attn - min) / max(max-min, 1e-8)
// Single fused kernel: bf16x3 split-precision WMMA GEMM -> in-LDS bitonic
// sort -> shuffle-scan threshold -> register epilogue.
//
// Tile mapping: 72 (=2x36) 16x16 tiles per 32-row block, 8 waves.
//   mt = w & 1, nt = (w>>1) + 4*i  (i = 0..8)
// => each wave touches ONE M-tile, so A hi/lo fragments are loaded once per
//    K-chunk (4x ds_load_b128) and reused across all 27 WMMAs of the chunk.
// ---------------------------------------------------------------------------

typedef __bf16 bf16;
typedef __bf16 v16bf __attribute__((ext_vector_type(16)));
typedef float  v8f   __attribute__((ext_vector_type(8)));

#define D_DIM   576
#define M_TILE  32          // rows per block
#define K_CHUNK 32          // K per LDS stage (one wmma K step)
#define LDSTRIDE 40         // padded bf16 row stride (80B, conflict-friendly)
#define SROW    1041        // padded f32 sort-row stride (bank spread)
#define NPAD    1024        // bitonic size
#define NWAVES  8
#define TILES_PER_WAVE 9    // 72 tiles / 8 waves

// ---- LDS byte layout (phase 1 aliases phase 2) ----
// phase1: As_hi[32*40] As_lo[32*40] Bs_hi[576*40] Bs_lo[576*40]  = 97,280 B
// phase2: sortbuf f32[32*1041] (133,248 B) + float4 stats[32]    = 133,760 B
// 2 workgroups x 133,760 B = 267,520 B < 320 KB  -> 16 waves per WGP.
#define SMEM_BYTES 133760

__device__ __forceinline__ float wave_incl_scan(float v, int lane) {
#pragma unroll
  for (int d = 1; d < 32; d <<= 1) {
    float n = __shfl_up(v, d, 32);
    if (lane >= d) v += n;
  }
  return v;
}

__device__ __forceinline__ float wave_sum(float v) {
#pragma unroll
  for (int d = 16; d > 0; d >>= 1) v += __shfl_xor(v, d, 32);
  return v;  // all lanes hold the total
}

// A fragment (16x32 bf16, MxK): lane = m(0..15) | khalf<<4.
// VGPR v0..3 hold K pairs {2v+8h}, v4..7 hold {16+2(v-4)+8h} -> two b128 loads.
__device__ __forceinline__ v16bf load_frag_a(const bf16* plane, int mt, int lane) {
  const int m = lane & 15;
  const int h = (lane >> 4) << 3;  // 0 or 8
  const bf16* rp = plane + (mt * 16 + m) * LDSTRIDE;
  union { uint4 q[2]; v16bf v; } u;
  u.q[0] = *(const uint4*)(rp + h);
  u.q[1] = *(const uint4*)(rp + 16 + h);
  return u.v;
}

// B fragment (32x16 bf16, KxN): lane = n(0..15) | khalf<<4; lane holds column n,
// K = 8*khalf..+7 and 16+8*khalf..+7 (same order as A) -> two b128 loads.
// LDS stores Bs[n][k] (W is row-major [n][k], i.e. already B = W^T column-major).
__device__ __forceinline__ v16bf load_frag_b(const bf16* plane, int nt, int lane) {
  const int n = lane & 15;
  const int h = (lane >> 4) << 3;  // 0 or 8 bf16
  const bf16* rp = plane + (nt * 16 + n) * LDSTRIDE + h;
  union { uint4 q[2]; v16bf v; } u;
  u.q[0] = *(const uint4*)(rp);
  u.q[1] = *(const uint4*)(rp + 16);
  return u.v;
}

__global__ void __launch_bounds__(256, 1)
flat_sparse_attention_kernel(const float* __restrict__ x,
                             const float* __restrict__ Wm,
                             const float* __restrict__ bias,
                             const float* __restrict__ temp,
                             float* __restrict__ out) {
  extern __shared__ char smem[];
  bf16* As_hi = (bf16*)smem;                       // 32*40
  bf16* As_lo = As_hi + M_TILE * LDSTRIDE;         // 32*40
  bf16* Bs_hi = As_lo + M_TILE * LDSTRIDE;         // 576*40
  bf16* Bs_lo = Bs_hi + D_DIM * LDSTRIDE;          // 576*40
  float*  sb    = (float*)smem;                    // phase2: 32 x SROW
  float4* stats = (float4*)(smem + M_TILE * SROW * sizeof(float));

  const int tid  = threadIdx.x;
  const int lane = tid & 31;
  const int w    = tid >> 5;
  const int mt   = w & 1;        // this wave's (single) M-tile
  const int ntb  = w >> 1;       // N-tile base; nt = ntb + 4*i
  const int row0 = blockIdx.x * M_TILE;

  v8f acc[TILES_PER_WAVE];
#pragma unroll
  for (int i = 0; i < TILES_PER_WAVE; i++)
#pragma unroll
    for (int e = 0; e < 8; e++) acc[i][e] = 0.0f;

  // ---------------- Phase 1: bf16x3 split-precision WMMA GEMM ----------------
  for (int k0 = 0; k0 < D_DIM; k0 += K_CHUNK) {
    __syncthreads();  // previous chunk's consumers done before overwrite

    // Stage A chunk: 32 rows x 32 k (coalesced), split f32 -> hi/lo bf16
    for (int idx = tid; idx < M_TILE * K_CHUNK; idx += 256) {
      const int r = idx >> 5, kk = idx & 31;
      const float v = x[(size_t)(row0 + r) * D_DIM + k0 + kk];
      const bf16 h = (bf16)v;
      As_hi[r * LDSTRIDE + kk] = h;
      As_lo[r * LDSTRIDE + kk] = (bf16)(v - (float)h);
    }
    // Stage B chunk: W[n][k0..k0+31] for all 576 n (L2-resident W)
    for (int idx = tid; idx < D_DIM * K_CHUNK; idx += 256) {
      const int n = idx >> 5, kk = idx & 31;
      const float v = Wm[(size_t)n * D_DIM + k0 + kk];
      const bf16 h = (bf16)v;
      Bs_hi[n * LDSTRIDE + kk] = h;
      Bs_lo[n * LDSTRIDE + kk] = (bf16)(v - (float)h);
    }
    __syncthreads();

    // A fragments: loaded ONCE per chunk, reused by all 9 tiles (27 wmma)
    const v16bf a_hi = load_frag_a(As_hi, mt, lane);
    const v16bf a_lo = load_frag_a(As_lo, mt, lane);

#pragma unroll
    for (int i = 0; i < TILES_PER_WAVE; i++) {
      const int nt = ntb + 4 * i;
      const v16bf b_hi = load_frag_b(Bs_hi, nt, lane);
      const v16bf b_lo = load_frag_b(Bs_lo, nt, lane);
      acc[i] = __builtin_amdgcn_wmma_f32_16x16x32_bf16(
          false, a_hi, false, b_lo, (short)0, acc[i], false, false);
      acc[i] = __builtin_amdgcn_wmma_f32_16x16x32_bf16(
          false, a_lo, false, b_hi, (short)0, acc[i], false, false);
      acc[i] = __builtin_amdgcn_wmma_f32_16x16x32_bf16(
          false, a_hi, false, b_hi, (short)0, acc[i], false, false);
    }
  }
  __syncthreads();  // all GEMM reads done; LDS now re-used as sort buffer

  // -------- Epilogue into registers + sort buffer (bias, temperature) --------
  const float invT = 1.0f / fmaxf(temp[0], 0.5f);
  const int mbase = mt * 16 + ((lane >> 4) << 3);  // C/D layout: m = v + 8*khalf
#pragma unroll
  for (int i = 0; i < TILES_PER_WAVE; i++) {
    const int nt  = ntb + 4 * i;
    const int col = nt * 16 + (lane & 15);
    const float bv = bias[col];
#pragma unroll
    for (int v = 0; v < 8; v++) {
      const float z = (acc[i][v] + bv) * invT;
      acc[i][v] = z;
      sb[(mbase + v) * SROW + col] = z;
    }
  }
  // pad 576..1023 with -inf so they sort to the tail (descending)
  const float NEG_INF = -__builtin_inff();
  for (int idx = tid; idx < M_TILE * (NPAD - D_DIM); idx += 256) {
    const int r = idx / (NPAD - D_DIM);
    const int c = D_DIM + idx % (NPAD - D_DIM);
    sb[r * SROW + c] = NEG_INF;
  }
  __syncthreads();

  // ---------------- Phase 2: bitonic sort (descending), 32 rows ----------------
  for (int k = 2; k <= NPAD; k <<= 1) {
    for (int j = k >> 1; j > 0; j >>= 1) {
      for (int p = tid; p < M_TILE * (NPAD / 2); p += 256) {
        const int row = p >> 9;
        const int q   = p & 511;
        const int i   = ((q & ~(j - 1)) << 1) | (q & (j - 1));
        const int l   = i + j;
        float* base = sb + row * SROW;
        const float a = base[i], bvl = base[l];
        const bool up   = ((i & k) == 0);           // descending region
        const bool swap = up ? (a < bvl) : (a > bvl);
        if (swap) { base[i] = bvl; base[l] = a; }
      }
      __syncthreads();
    }
  }

  // ------- Per-row threshold: cumsum scan, k_max, tau, amin/amax/denom -------
  for (int rb = 0; rb < 4; rb++) {
    const int r = w * 4 + rb;
    const float* zs = sb + r * SROW;
    float run = 0.0f, cnt = 0.0f;
    for (int c = 0; c < 18; c++) {
      const int i = c * 32 + lane;
      const float z = zs[i];
      const float s = wave_incl_scan(z, lane);
      const float cum = run + s;                           // raw cumsum
      const float th  = (cum - 2.0f) / (float)(i + 1);     // (cumsum-1-1)/k
      cnt += (th >= z) ? 1.0f : 0.0f;
      run += __shfl(s, 31, 32);
    }
    int idx = (int)wave_sum(cnt) - 1;                      // k_max - 1
    if (idx < 0) idx = 0;
    float part = 0.0f;
    for (int c = 0; c < 18; c++) {
      const int i = c * 32 + lane;
      part += (i <= idx) ? zs[i] : 0.0f;
    }
    const float tau  = (wave_sum(part) - 2.0f) / (float)(idx + 1);
    const float amax = fmaxf(zs[0]   - tau, 0.0f);   // sorted[0]  = row max
    const float amin = fmaxf(zs[575] - tau, 0.0f);   // sorted[575]= row min
    const float denom = fmaxf(amax - amin, 1e-8f);
    if (lane == 0) stats[r] = make_float4(tau, amin, denom, 0.0f);
  }
  __syncthreads();

  // ---------------- Output from registers (coalesced f32 store) ----------------
#pragma unroll
  for (int i = 0; i < TILES_PER_WAVE; i++) {
    const int nt  = ntb + 4 * i;
    const int col = nt * 16 + (lane & 15);
#pragma unroll
    for (int v = 0; v < 8; v++) {
      const int r = mbase + v;
      const float4 st = stats[r];  // {tau, amin, denom}
      const float a = fmaxf(acc[i][v] - st.x, 0.0f);
      out[(size_t)(row0 + r) * D_DIM + col] = (a - st.y) / st.z;
    }
  }
}

extern "C" void kernel_launch(void* const* d_in, const int* in_sizes, int n_in,
                              void* d_out, int out_size, void* d_ws, size_t ws_size,
                              hipStream_t stream) {
  (void)in_sizes; (void)n_in; (void)d_ws; (void)ws_size; (void)out_size;
  const float* x    = (const float*)d_in[0];
  const float* Wm   = (const float*)d_in[1];
  const float* bias = (const float*)d_in[2];
  const float* temp = (const float*)d_in[3];
  float* out = (float*)d_out;

  hipFuncSetAttribute((const void*)flat_sparse_attention_kernel,
                      hipFuncAttributeMaxDynamicSharedMemorySize, SMEM_BYTES);

  const int nblocks = 65536 / M_TILE;  // 2048
  flat_sparse_attention_kernel<<<nblocks, 256, SMEM_BYTES, stream>>>(
      x, Wm, bias, temp, out);
}
